// EnergyModelT_67061619360160
// MI455X (gfx1250) — matrix operand order repaired and checked by reading, not compiled, not run on hardware
//
#include <hip/hip_runtime.h>
#include <hip/hip_bf16.h>
#include <math.h>

typedef unsigned short u16;
typedef __attribute__((ext_vector_type(16))) __bf16 v16bf;
typedef __attribute__((ext_vector_type(8)))  float  v8f;

#define N_ATOMS 100000
#define N_EDGES 1600000
#define NRAD    16
#define HIDDEN  512
#define NFEAT   272
#define KPAD1   288            // 272 padded to multiple of 32 for WMMA K-steps
#define NPAD    100096         // N_ATOMS rounded up to multiple of 128
#define EPSV    1e-8f

__device__ __forceinline__ u16 f2bf(float f) {
  unsigned int u = __float_as_uint(f);
  u += 0x7FFFu + ((u >> 16) & 1u);       // round-to-nearest-even
  return (u16)(u >> 16);
}
__device__ __forceinline__ float bf2f(u16 h) {
  return __uint_as_float(((unsigned int)h) << 16);
}

// ---------------------------------------------------------------------------
// Convert W1 (272x512) and W2 (512x512) f32 row-major into bf16, transposed so
// each output column's K-run is contiguous: Wt[n][k], k contiguous, K padded.
// A 16-column tile of Wt is then a single contiguous 16*K*2-byte block, which
// is exactly what the async LDS copy engine wants.
// ---------------------------------------------------------------------------
__global__ __launch_bounds__(256) void prep_weights(
    const float* __restrict__ W1, const float* __restrict__ W2,
    u16* __restrict__ Wt1, u16* __restrict__ Wt2)
{
  int t = blockIdx.x * blockDim.x + threadIdx.x;
  const int total1 = HIDDEN * KPAD1;
  if (t < total1) {
    int n = t / KPAD1, k = t - n * KPAD1;
    Wt1[t] = (k < NFEAT) ? f2bf(W1[(size_t)k * HIDDEN + n]) : (u16)0;
  } else {
    int t2 = t - total1;
    if (t2 < HIDDEN * HIDDEN) {
      int n = t2 / HIDDEN, k = t2 - n * HIDDEN;
      Wt2[t2] = f2bf(W2[(size_t)k * HIDDEN + n]);
    }
  }
}

// ---------------------------------------------------------------------------
// Edge phase: Gaussian radial basis + scatter-add moments (atomics into L2-
// resident 25.6MB accumulator region).
// ---------------------------------------------------------------------------
__global__ __launch_bounds__(256) void edge_kernel(
    const float* __restrict__ R, const int* __restrict__ idx,
    const float* __restrict__ centers, const float* __restrict__ width,
    float* __restrict__ M0, float* __restrict__ M1)
{
  int e = blockIdx.x * blockDim.x + threadIdx.x;
  if (e >= N_EDGES) return;
  int i = idx[e];
  int j = idx[N_EDGES + e];
  float dx = R[3 * j + 0] - R[3 * i + 0];
  float dy = R[3 * j + 1] - R[3 * i + 1];
  float dz = R[3 * j + 2] - R[3 * i + 2];
  float r  = sqrtf(dx * dx + dy * dy + dz * dz);
  float inv = 1.0f / (r + EPSV);
  float ux = dx * inv, uy = dy * inv, uz = dz * inv;
  float w = width[0];
  float inv2w2 = 1.0f / (2.0f * w * w);
  float* m0 = M0 + (size_t)i * NRAD;
  float* m1 = M1 + (size_t)i * NRAD * 3;
#pragma unroll
  for (int k = 0; k < NRAD; ++k) {
    float t = r - centers[k];
    float b = __expf(-t * t * inv2w2);
    atomicAdd(m0 + k, b);
    atomicAdd(m1 + 3 * k + 0, b * ux);
    atomicAdd(m1 + 3 * k + 1, b * uy);
    atomicAdd(m1 + 3 * k + 2, b * uz);
  }
}

// ---------------------------------------------------------------------------
// Per-atom features: feat = [M0, M1 M1^T] as bf16, row length KPAD1 (padded).
// ---------------------------------------------------------------------------
__global__ __launch_bounds__(256) void feat_kernel(
    const float* __restrict__ M0, const float* __restrict__ M1,
    u16* __restrict__ feat)
{
  int i = blockIdx.x * blockDim.x + threadIdx.x;
  if (i >= NPAD) return;
  u16* frow = feat + (size_t)i * KPAD1;
  if (i >= N_ATOMS) {
#pragma unroll 4
    for (int k = 0; k < KPAD1; ++k) frow[k] = 0;
    return;
  }
  const float* m1 = M1 + (size_t)i * NRAD * 3;
  float m1x[NRAD], m1y[NRAD], m1z[NRAD];
#pragma unroll
  for (int k = 0; k < NRAD; ++k) {
    m1x[k] = m1[3 * k + 0];
    m1y[k] = m1[3 * k + 1];
    m1z[k] = m1[3 * k + 2];
  }
  const float* m0 = M0 + (size_t)i * NRAD;
#pragma unroll
  for (int k = 0; k < NRAD; ++k) frow[k] = f2bf(m0[k]);
#pragma unroll
  for (int a = 0; a < NRAD; ++a) {
#pragma unroll
    for (int b = 0; b < NRAD; ++b) {
      float g = m1x[a] * m1x[b] + m1y[a] * m1y[b] + m1z[a] * m1z[b];
      frow[NRAD + a * NRAD + b] = f2bf(g);
    }
  }
#pragma unroll
  for (int k = NFEAT; k < KPAD1; ++k) frow[k] = 0;
}

// ---------------------------------------------------------------------------
// Fused GEMM + bias + SiLU using v_wmma_f32_16x16x32_bf16 with async-LDS
// double-buffered weight staging and a register-pipelined B-fragment stream.
//
// Block = 256 threads = 8 wave32s; block tile = 128 rows. Wave w owns rows
// [row0+16w, row0+16w+16). Weight column tiles (16 cols x KP, contiguous in
// the transposed layout) are staged into LDS with
// GLOBAL_LOAD_ASYNC_TO_LDS_B128 (ASYNCcnt), double-buffered so the copy of
// tile nt+1 overlaps the WMMA sweep of tile nt; all 8 waves share each tile.
// Inside the K-loop the B fragment for step kt+1 is ds_load'ed before the
// WMMA consuming step kt, so the scheduler can use s_wait_dscnt 2 instead of
// a full LDS drain before every matrix op.
//
// A fragment (16x32 bf16, contiguous K order, same as B): lane m=lane&15, kh=lane>>4;
//   elems 0..7  <- K = k0 + kh*16 + e       (16B contiguous load)
//   elems 8..15 <- K = k0 + kh*16 + 8 + e   (16B contiguous load)
// B fragment (32x16): lane holds column n=nt*16+m, K-run k0+kh*16 .. +15,
//   32 contiguous bytes in LDS (ds_load_b128 x2).
// C/D: VGPR r -> M = r + 8*kh, N = lane&15.
// ---------------------------------------------------------------------------
template <int KP>
__global__ __launch_bounds__(256) void gemm_bias_silu(
    const u16* __restrict__ A, const u16* __restrict__ Wt,
    const float* __restrict__ bias, u16* __restrict__ out)
{
  constexpr int NKT      = KP / 32;          // WMMA K-steps
  constexpr int TILE_U16 = 16 * KP;          // one 16-column weight tile
  constexpr int NCHUNK   = 2 * KP;           // 16B chunks per tile (16*KP*2/16)
  constexpr int NCOPY    = (NCHUNK + 255) / 256;  // async copies per thread
  constexpr int KWAIT    = KP / 128;         // per-wave safe wait constant

  __shared__ u16 bbuf[2 * TILE_U16];         // double buffer (18KB / 32KB)

  const int tid  = threadIdx.x;
  const int lane = tid & 31;
  const int wave = tid >> 5;
  const int row0 = blockIdx.x * 128 + wave * 16;
  const int m  = lane & 15;
  const int kh = lane >> 4;

  // ---- preload A fragments (reused across all 32 N-tiles) ----
  const u16* Arow = A + (size_t)(row0 + m) * KP;
  v16bf afrag[NKT];
#pragma unroll
  for (int kt = 0; kt < NKT; ++kt) {
    union { v16bf v; uint4 u[2]; } fa;
    fa.u[0] = *(const uint4*)(Arow + kt * 32 + kh * 16);
    fa.u[1] = *(const uint4*)(Arow + kt * 32 + kh * 16 + 8);
    afrag[kt] = fa.v;
  }

  // LDS byte offsets of the two staging buffers (addr[31:0] of flat address
  // is the LDS offset per the aperture mapping, ISA 10.2)
  const unsigned int ldsBase0 = (unsigned int)(uintptr_t)(&bbuf[0]);
  const unsigned int ldsBase1 = (unsigned int)(uintptr_t)(&bbuf[TILE_U16]);

  // ---- async prefetch of tile 0 into buffer 0 ----
  {
    const u16* src = Wt;                     // tile 0 is at Wt + 0
#pragma unroll
    for (int c = 0; c < NCOPY; ++c) {
      int ch = tid + c * 256;
      if (ch < NCHUNK) {
        unsigned int go = (unsigned int)ch * 16u;
        asm volatile("global_load_async_to_lds_b128 %0, %1, %2"
                     :: "v"(ldsBase0 + go), "v"(go), "s"(src)
                     : "memory");
      }
    }
  }

  for (int nt = 0; nt < 32; ++nt) {
    const int cur = nt & 1;
    // ---- prefetch next tile into the other buffer ----
    if (nt + 1 < 32) {
      const u16* src = Wt + (size_t)(nt + 1) * TILE_U16;
      const unsigned int dstBase = (cur ? ldsBase0 : ldsBase1);
#pragma unroll
      for (int c = 0; c < NCOPY; ++c) {
        int ch = tid + c * 256;
        if (ch < NCHUNK) {
          unsigned int go = (unsigned int)ch * 16u;
          asm volatile("global_load_async_to_lds_b128 %0, %1, %2"
                       :: "v"(dstBase + go), "v"(go), "s"(src)
                       : "memory");
        }
      }
      // current tile's copies are done once outstanding <= this wave's
      // freshly-issued prefetch count (async ops complete in order)
      asm volatile("s_wait_asynccnt %0" :: "i"(KWAIT) : "memory");
    } else {
      asm volatile("s_wait_asynccnt 0" ::: "memory");
    }
    __syncthreads();                         // all waves' copies visible

    // ---- WMMA sweep of current tile from LDS (B pipelined one step ahead) --
    const u16* Blds = &bbuf[cur * TILE_U16 + m * KP + kh * 16];
    v8f acc = {};
    v16bf bcur = *(const v16bf*)(Blds);
#pragma unroll
    for (int kt = 0; kt < NKT; ++kt) {
      v16bf bnext = bcur;
      if (kt + 1 < NKT) bnext = *(const v16bf*)(Blds + (kt + 1) * 32);
      acc = __builtin_amdgcn_wmma_f32_16x16x32_bf16(
          /*neg_a=*/false, afrag[kt], /*neg_b=*/false, bcur,
          /*c_mod=*/(short)0, acc, /*reuse_a=*/false, /*reuse_b=*/false);
      bcur = bnext;
    }

    const int col = nt * 16 + m;
    const float bc = bias[col];
#pragma unroll
    for (int r = 0; r < 8; ++r) {
      float v = acc[r] + bc;
      float s = v / (1.0f + __expf(-v));     // SiLU
      out[(size_t)(row0 + r + kh * 8) * HIDDEN + col] = f2bf(s);
    }
    __syncthreads();                         // buffer reuse protection
  }
}

// ---------------------------------------------------------------------------
// Layer 3 (512 -> 1) + per-element scale/shift + global energy reduction.
// ---------------------------------------------------------------------------
__global__ __launch_bounds__(256) void readout_kernel(
    const u16* __restrict__ h2, const float* __restrict__ W3,
    const float* __restrict__ b3, const int* __restrict__ Z,
    const float* __restrict__ scale, const float* __restrict__ shift,
    float* __restrict__ out)
{
  __shared__ float red[256];
  int i = blockIdx.x * blockDim.x + threadIdx.x;
  float e = 0.0f;
  if (i < N_ATOMS) {
    const u16* hrow = h2 + (size_t)i * HIDDEN;
    float acc = 0.0f;
#pragma unroll 8
    for (int k = 0; k < HIDDEN; ++k) acc += bf2f(hrow[k]) * W3[k];
    acc += b3[0];
    int z = Z[i];
    e = scale[z] * acc + shift[z];
  }
  red[threadIdx.x] = e;
  __syncthreads();
  for (int s = 128; s > 0; s >>= 1) {
    if ((int)threadIdx.x < s) red[threadIdx.x] += red[threadIdx.x + s];
    __syncthreads();
  }
  if (threadIdx.x == 0) atomicAdd(out, red[0]);
}

// ---------------------------------------------------------------------------
extern "C" void kernel_launch(void* const* d_in, const int* in_sizes, int n_in,
                              void* d_out, int out_size, void* d_ws, size_t ws_size,
                              hipStream_t stream)
{
  const float* R       = (const float*)d_in[0];
  const int*   Z       = (const int*)  d_in[1];
  const int*   idx     = (const int*)  d_in[2];
  // d_in[3] = box (zeros), d_in[4] = offsets (zeros) -> free displacement
  const float* centers = (const float*)d_in[5];
  const float* width   = (const float*)d_in[6];
  const float* W1      = (const float*)d_in[7];
  const float* b1      = (const float*)d_in[8];
  const float* W2      = (const float*)d_in[9];
  const float* b2      = (const float*)d_in[10];
  const float* W3      = (const float*)d_in[11];
  const float* b3      = (const float*)d_in[12];
  const float* scale   = (const float*)d_in[13];
  const float* shift   = (const float*)d_in[14];

  char* ws = (char*)d_ws;
  size_t off = 0;
  float* M0 = (float*)(ws + off); off += (size_t)N_ATOMS * NRAD * sizeof(float);
  float* M1 = (float*)(ws + off); off += (size_t)N_ATOMS * NRAD * 3 * sizeof(float);
  size_t momBytes = off;
  u16* feat = (u16*)(ws + off); off += (size_t)NPAD * KPAD1  * sizeof(u16);
  u16* h1   = (u16*)(ws + off); off += (size_t)NPAD * HIDDEN * sizeof(u16);
  u16* h2   = (u16*)(ws + off); off += (size_t)NPAD * HIDDEN * sizeof(u16);
  u16* Wt1  = (u16*)(ws + off); off += (size_t)HIDDEN * KPAD1  * sizeof(u16);
  u16* Wt2  = (u16*)(ws + off); off += (size_t)HIDDEN * HIDDEN * sizeof(u16);

  hipMemsetAsync(d_ws, 0, momBytes, stream);          // zero moment accumulators
  hipMemsetAsync(d_out, 0, sizeof(float), stream);    // zero energy accumulator

  const int wtot = HIDDEN * KPAD1 + HIDDEN * HIDDEN;
  prep_weights<<<(wtot + 255) / 256, 256, 0, stream>>>(W1, W2, Wt1, Wt2);
  edge_kernel<<<(N_EDGES + 255) / 256, 256, 0, stream>>>(R, idx, centers, width, M0, M1);
  feat_kernel<<<(NPAD + 255) / 256, 256, 0, stream>>>(M0, M1, feat);
  gemm_bias_silu<KPAD1 ><<<NPAD / 128, 256, 0, stream>>>(feat, Wt1, b1, h1);
  gemm_bias_silu<HIDDEN><<<NPAD / 128, 256, 0, stream>>>(h1,   Wt2, b2, h2);
  readout_kernel<<<(N_ATOMS + 255) / 256, 256, 0, stream>>>(h2, W3, b3, Z, scale, shift,
                                                            (float*)d_out);
}
